// HOIST_25142738551207
// MI455X (gfx1250) — matrix operand order repaired and checked by reading, not compiled, not run on hardware
//
#include <hip/hip_runtime.h>
#include <hip/hip_bf16.h>
#include <stdint.h>

#define NN   2000
#define TT   64
#define HH   128
#define KK   4
#define FF   64          // total dynamic feature dim (16+32+16)
#define G4   512         // 4*H
#define KPAD 2016        // 2000 padded up to 63*32 for WMMA K loop
#define COLS 8192        // T*H
#define MPAD 2048        // M padded so 64-row workgroup blocking divides evenly

typedef __attribute__((ext_vector_type(16))) _Float16 v16h;
typedef __attribute__((ext_vector_type(8)))  _Float16 v8h;
typedef __attribute__((ext_vector_type(8)))  float    v8f;

__device__ __forceinline__ float sigm(float x)  { return 1.0f / (1.0f + expf(-x)); }
__device__ __forceinline__ float lrelu(float x) { return x > 0.0f ? x : 0.01f * x; }

__device__ __forceinline__ v16h combine16(v8h x0, v8h x1) {
  v16h r;
#pragma unroll
  for (int e = 0; e < 8; e++) { r[e] = x0[e]; r[e + 8] = x1[e]; }
  return r;
}

// ---------------- WMMA fragment loaders (wave32, v_wmma_f32_16x16x32_f16) ----------------
// A (16x32 f16, row-major, tile base = &A[m0][0], row stride lda in halves):
//   lane L: m = L%16, kh = L/16; halves e=0..15 -> k = 16*kh + e  (contiguous)
__device__ __forceinline__ v16h load_a(const _Float16* Atile, int lda, int lane, int k0) {
  int m  = lane & 15;
  int kh = (lane >> 4) & 1;
  const _Float16* p = Atile + (size_t)m * lda + k0 + 16 * kh;
  return combine16(*(const v8h*)(p), *(const v8h*)(p + 8));
}
// B (32x16 f16) supplied pre-transposed as Bp (Ncols x K row-major, tile base = &Bp[n0][0]):
//   lane L: n = L%16, kg = L/16; halves e=0..15 -> k = 16*kg + e  (contiguous)
__device__ __forceinline__ v16h load_b(const _Float16* Bptile, int ldb, int lane, int k0) {
  int n  = lane & 15;
  int kg = (lane >> 4) & 1;
  const _Float16* p = Bptile + (size_t)n * ldb + k0 + 16 * kg;
  return combine16(*(const v8h*)(p), *(const v8h*)(p + 8));
}

// ---------------- prep: f16 weight conversion, bias fold, state zeroing, padding ----------------
__global__ void prep_kernel(const float* __restrict__ Wih, const float* __restrict__ Whh,
                            const float* __restrict__ bih, const float* __restrict__ bhh,
                            _Float16* __restrict__ WihH, _Float16* __restrict__ WhhH,
                            float* __restrict__ bsum, float* __restrict__ c_state,
                            _Float16* __restrict__ h16, _Float16* __restrict__ hT,
                            _Float16* __restrict__ Adis) {
  int i = blockIdx.x * blockDim.x + threadIdx.x;
  if (i < G4 * FF) WihH[i] = (_Float16)Wih[i];
  if (i < G4 * HH) WhhH[i] = (_Float16)Whh[i];
  if (i < G4)      bsum[i] = bih[i] + bhh[i];
  if (i < NN * HH) { c_state[i] = 0.0f; h16[i] = (_Float16)0.0f; }
  if (i < COLS * 16) {                       // zero the K-padding columns of hT
    int r = i >> 4; int cc = NN + (i & 15);
    hT[(size_t)r * KPAD + cc] = (_Float16)0.0f;
  }
  if (i < (MPAD - NN) * KPAD) {              // zero the M-padding rows of Adis
    int r = NN + i / KPAD; int cc = i % KPAD;
    Adis[(size_t)r * KPAD + cc] = (_Float16)0.0f;
  }
}

// ---------------- per-node GAT score terms ----------------
__global__ void score_kernel(const float* __restrict__ st0, const float* __restrict__ w0,
                             const float* __restrict__ a0,
                             const float* __restrict__ st1, const float* __restrict__ w1,
                             const float* __restrict__ a1,
                             float* __restrict__ s0a, float* __restrict__ s0b,
                             float* __restrict__ s1a, float* __restrict__ s1b) {
  int n = blockIdx.x * blockDim.x + threadIdx.x;
  if (n >= NN) return;
  {
    const int d = 8; float hi[8];
#pragma unroll
    for (int j = 0; j < d; j++) {
      float s = 0.0f;
      for (int i = 0; i < d; i++) s += st0[n * d + i] * w0[i * d + j];
      hi[j] = s;
    }
    float sa = 0.0f, sb = 0.0f;
#pragma unroll
    for (int j = 0; j < d; j++) { sa += hi[j] * a0[j]; sb += hi[j] * a0[d + j]; }
    s0a[n] = sa; s0b[n] = sb;
  }
  {
    const int d = 16; float hi[16];
#pragma unroll
    for (int j = 0; j < d; j++) {
      float s = 0.0f;
      for (int i = 0; i < d; i++) s += st1[n * d + i] * w1[i * d + j];
      hi[j] = s;
    }
    float sa = 0.0f, sb = 0.0f;
#pragma unroll
    for (int j = 0; j < d; j++) { sa += hi[j] * a1[j]; sb += hi[j] * a1[d + j]; }
    s1a[n] = sa; s1b[n] = sb;
  }
}

// ---------------- pairwise logits + row softmax (+I folded in) -> Adis f16 (MPAD x KPAD) ----------------
__global__ void attn_kernel(const float* __restrict__ s0a, const float* __restrict__ s0b,
                            const float* __restrict__ s1a, const float* __restrict__ s1b,
                            const float* __restrict__ dist, const float* __restrict__ Wdis,
                            const float* __restrict__ adis, _Float16* __restrict__ Adis) {
  __shared__ float logits[NN];
  __shared__ float red[256];
  int n = blockIdx.x, tid = threadIdx.x;
  float v0 = 0, v1 = 0, v2 = 0, v3 = 0;
#pragma unroll
  for (int j = 0; j < KK; j++) {
    float aj = adis[j];
    v0 += Wdis[0 * KK + j] * aj; v1 += Wdis[1 * KK + j] * aj;
    v2 += Wdis[2 * KK + j] * aj; v3 += Wdis[3 * KK + j] * aj;
  }
  float va = s0a[n], vb = s1a[n];
  float lmax = -1e30f;
  for (int m = tid; m < NN; m += 256) {
    const float* dp = dist + ((size_t)n * NN + m) * KK;
    float dd = dp[0] * v0 + dp[1] * v1 + dp[2] * v2 + dp[3] * v3;
    float l = sigm(va + s0b[m]) + sigm(vb + s1b[m]) + sigm(dd);
    logits[m] = l;
    lmax = fmaxf(lmax, l);
  }
  red[tid] = lmax; __syncthreads();
  for (int s = 128; s > 0; s >>= 1) { if (tid < s) red[tid] = fmaxf(red[tid], red[tid + s]); __syncthreads(); }
  float mx = red[0]; __syncthreads();
  float lsum = 0.0f;
  for (int m = tid; m < NN; m += 256) { float e = expf(logits[m] - mx); logits[m] = e; lsum += e; }
  red[tid] = lsum; __syncthreads();
  for (int s = 128; s > 0; s >>= 1) { if (tid < s) red[tid] += red[tid + s]; __syncthreads(); }
  float inv = 1.0f / red[0];
  for (int m = tid; m < NN; m += 256) {
    float p = logits[m] * inv;
    if (m == n) p += 1.0f;                 // fold the residual identity into the GEMM
    Adis[(size_t)n * KPAD + m] = (_Float16)p;
  }
  for (int m = NN + tid; m < KPAD; m += 256) Adis[(size_t)n * KPAD + m] = (_Float16)0.0f;
}

// ---------------- gating MLPs: x = sigmoid(lrelu(dyn@W1+b1)@W2+b2) * dyn -> f16 (N,T,64) ----------------
__global__ void gating_kernel(const float* __restrict__ dyn0, const float* __restrict__ W1_0,
                              const float* __restrict__ b1_0, const float* __restrict__ W2_0,
                              const float* __restrict__ b2_0,
                              const float* __restrict__ dyn1, const float* __restrict__ W1_1,
                              const float* __restrict__ b1_1, const float* __restrict__ W2_1,
                              const float* __restrict__ b2_1,
                              const float* __restrict__ dyn2, const float* __restrict__ W1_2,
                              const float* __restrict__ b1_2, const float* __restrict__ W2_2,
                              const float* __restrict__ b2_2,
                              _Float16* __restrict__ xout) {
  __shared__ float sdyn[FF];
  __shared__ float shid[HH];
  int nt = blockIdx.x;
  int tid = threadIdx.x;                // 128 threads
  size_t row = (size_t)nt;
  if (tid < 16) sdyn[tid]      = dyn0[row * 16 + tid];
  if (tid < 32) sdyn[16 + tid] = dyn1[row * 32 + tid];
  if (tid < 16) sdyn[48 + tid] = dyn2[row * 16 + tid];
  __syncthreads();
  const float* W1a[3] = { W1_0, W1_1, W1_2 };
  const float* b1a[3] = { b1_0, b1_1, b1_2 };
  const float* W2a[3] = { W2_0, W2_1, W2_2 };
  const float* b2a[3] = { b2_0, b2_1, b2_2 };
  const int dims[3] = { 16, 32, 16 };
  const int offs[3] = { 0, 16, 48 };
  for (int cat = 0; cat < 3; cat++) {
    int d = dims[cat], off = offs[cat];
    float hj = b1a[cat][tid];
    for (int i = 0; i < d; i++) hj += sdyn[off + i] * W1a[cat][i * HH + tid];
    shid[tid] = lrelu(hj);
    __syncthreads();
    if (tid < d) {
      float o = b2a[cat][tid];
      for (int j = 0; j < HH; j++) o += shid[j] * W2a[cat][j * d + tid];
      xout[row * FF + off + tid] = (_Float16)(sigm(o) * sdyn[off + tid]);
    }
    __syncthreads();
  }
}

// ---------------- LSTM step GEMM (WMMA, 1x4 N-blocked): G = x_t@Wih^T + h@Whh^T + bias ----------------
__global__ __launch_bounds__(128)
void lstm_gemm_kernel(const _Float16* __restrict__ x, int t,
                      const _Float16* __restrict__ hprev,
                      const _Float16* __restrict__ WihH,
                      const _Float16* __restrict__ WhhH,
                      const float* __restrict__ bsum,
                      float* __restrict__ Gm) {
  const int NTM = NN / 16, NGN = G4 / 64;          // 125 x 8 wave-tiles (16 x 64 each)
  int wid = blockIdx.x * (blockDim.x >> 5) + (threadIdx.x >> 5);
  if (wid >= NTM * NGN) return;
  int tm = wid / NGN, tg = wid % NGN;
  int lane = threadIdx.x & 31;
  int m0 = tm * 16, n0 = tg * 64;
  v8f c0 = {}, c1 = {}, c2 = {}, c3 = {};
  const _Float16* Ax = x + ((size_t)m0 * TT + t) * FF;   // row stride T*F
#pragma unroll
  for (int k0 = 0; k0 < FF; k0 += 32) {
    v16h a = load_a(Ax, TT * FF, lane, k0);
    v16h b;
    b = load_b(WihH + (size_t)(n0 +  0) * FF, FF, lane, k0);
    c0 = __builtin_amdgcn_wmma_f32_16x16x32_f16(false, a, false, b, (short)0, c0, false, false);
    b = load_b(WihH + (size_t)(n0 + 16) * FF, FF, lane, k0);
    c1 = __builtin_amdgcn_wmma_f32_16x16x32_f16(false, a, false, b, (short)0, c1, false, false);
    b = load_b(WihH + (size_t)(n0 + 32) * FF, FF, lane, k0);
    c2 = __builtin_amdgcn_wmma_f32_16x16x32_f16(false, a, false, b, (short)0, c2, false, false);
    b = load_b(WihH + (size_t)(n0 + 48) * FF, FF, lane, k0);
    c3 = __builtin_amdgcn_wmma_f32_16x16x32_f16(false, a, false, b, (short)0, c3, false, false);
  }
  const _Float16* Ah = hprev + (size_t)m0 * HH;
#pragma unroll
  for (int k0 = 0; k0 < HH; k0 += 32) {
    v16h a = load_a(Ah, HH, lane, k0);
    v16h b;
    b = load_b(WhhH + (size_t)(n0 +  0) * HH, HH, lane, k0);
    c0 = __builtin_amdgcn_wmma_f32_16x16x32_f16(false, a, false, b, (short)0, c0, false, false);
    b = load_b(WhhH + (size_t)(n0 + 16) * HH, HH, lane, k0);
    c1 = __builtin_amdgcn_wmma_f32_16x16x32_f16(false, a, false, b, (short)0, c1, false, false);
    b = load_b(WhhH + (size_t)(n0 + 32) * HH, HH, lane, k0);
    c2 = __builtin_amdgcn_wmma_f32_16x16x32_f16(false, a, false, b, (short)0, c2, false, false);
    b = load_b(WhhH + (size_t)(n0 + 48) * HH, HH, lane, k0);
    c3 = __builtin_amdgcn_wmma_f32_16x16x32_f16(false, a, false, b, (short)0, c3, false, false);
  }
  int nl = lane & 15;
  int mb = m0 + ((lane >> 4) ? 8 : 0);
#pragma unroll
  for (int r = 0; r < 8; r++) {
    Gm[(size_t)(mb + r) * G4 + n0 +  0 + nl] = c0[r] + bsum[n0 +  0 + nl];
    Gm[(size_t)(mb + r) * G4 + n0 + 16 + nl] = c1[r] + bsum[n0 + 16 + nl];
    Gm[(size_t)(mb + r) * G4 + n0 + 32 + nl] = c2[r] + bsum[n0 + 32 + nl];
    Gm[(size_t)(mb + r) * G4 + n0 + 48 + nl] = c3[r] + bsum[n0 + 48 + nl];
  }
}

// ---------------- LSTM gate nonlinearity; writes next-state f16 and transposed h-sequence ----------------
__global__ void lstm_gate_kernel(const float* __restrict__ Gm, float* __restrict__ c_state,
                                 _Float16* __restrict__ h16, _Float16* __restrict__ hT, int t) {
  int idx = blockIdx.x * blockDim.x + threadIdx.x;
  if (idx >= NN * HH) return;
  int n = idx / HH, d = idx % HH;
  size_t gb = (size_t)n * G4;
  float i_ = Gm[gb + d];
  float f_ = Gm[gb + HH + d];
  float g_ = Gm[gb + 2 * HH + d];
  float o_ = Gm[gb + 3 * HH + d];
  float c = sigm(f_) * c_state[idx] + sigm(i_) * tanhf(g_);
  float h = sigm(o_) * tanhf(c);
  c_state[idx] = c;
  h16[idx] = (_Float16)h;
  hT[(size_t)(t * HH + d) * KPAD + n] = (_Float16)h;
}

// ---------------- spatial aggregation (WMMA, 64x64 block tile, async B staging via LDS) ----------------
// hagg = (softmax + I) @ h : (2048x2016) @ (2016x8192); B tiles shared by 4 waves through LDS,
// staged with global_load_async_to_lds_b128 (ASYNCcnt), A fragments reused 4x in registers.
__global__ __launch_bounds__(128)
void agg_gemm_kernel(const _Float16* __restrict__ Adis, const _Float16* __restrict__ hT,
                     float* __restrict__ hagg) {
  // 64 rows x 40 halves (80B pitch -> ds_load_b128 reads are bank-conflict-free)
  __shared__ _Float16 sB[64 * 40];
  const int NB = COLS / 64;                 // 128 n-groups
  int bm = blockIdx.x / NB;                 // 0..31  (MPAD/64)
  int bn = blockIdx.x % NB;
  int tid = threadIdx.x;
  int wave = tid >> 5;
  int lane = tid & 31;
  int m0 = bm * 64 + wave * 16;
  int n0 = bn * 64;

  // staging: thread stages 32B of B: row = tid>>1 (0..63), 32B chunk = tid&1
  int srow = tid >> 1, schunk = tid & 1;
  const _Float16* gB = hT + (size_t)(n0 + srow) * KPAD + schunk * 16;
  unsigned ldsb = (unsigned)(size_t)(sB + srow * 40 + schunk * 16);  // LDS byte offset (low 32 bits)

  const _Float16* At = Adis + (size_t)m0 * KPAD;
  int nl = lane & 15;
  int kg = (lane >> 4) & 1;
  const _Float16* pb0 = sB + kg * 16 + (size_t)nl * 40;
  v8f c0 = {}, c1 = {}, c2 = {}, c3 = {};

  for (int k0 = 0; k0 < KPAD; k0 += 32) {
    const _Float16* gp = gB + k0;
    asm volatile("global_load_async_to_lds_b128 %0, %1, off"
                 :: "v"(ldsb), "v"(gp) : "memory");
    asm volatile("global_load_async_to_lds_b128 %0, %1, off offset:16"
                 :: "v"(ldsb), "v"(gp) : "memory");
    if (k0 + 32 < KPAD)
      __builtin_prefetch((const void*)(At + (size_t)nl * KPAD + k0 + 32), 0, 1);
    asm volatile("s_wait_asynccnt 0x0" ::: "memory");
    __syncthreads();

    v16h a = load_a(At, KPAD, lane, k0);
    v16h b;
    b = combine16(*(const v8h*)(pb0 +  0 * 40), *(const v8h*)(pb0 +  0 * 40 + 8));
    c0 = __builtin_amdgcn_wmma_f32_16x16x32_f16(false, a, false, b, (short)0, c0, false, false);
    b = combine16(*(const v8h*)(pb0 + 16 * 40), *(const v8h*)(pb0 + 16 * 40 + 8));
    c1 = __builtin_amdgcn_wmma_f32_16x16x32_f16(false, a, false, b, (short)0, c1, false, false);
    b = combine16(*(const v8h*)(pb0 + 32 * 40), *(const v8h*)(pb0 + 32 * 40 + 8));
    c2 = __builtin_amdgcn_wmma_f32_16x16x32_f16(false, a, false, b, (short)0, c2, false, false);
    b = combine16(*(const v8h*)(pb0 + 48 * 40), *(const v8h*)(pb0 + 48 * 40 + 8));
    c3 = __builtin_amdgcn_wmma_f32_16x16x32_f16(false, a, false, b, (short)0, c3, false, false);
    __syncthreads();
  }

  int mb = m0 + ((lane >> 4) ? 8 : 0);
#pragma unroll
  for (int r = 0; r < 8; r++) {
    hagg[(size_t)(mb + r) * COLS + n0 +  0 + nl] = c0[r];
    hagg[(size_t)(mb + r) * COLS + n0 + 16 + nl] = c1[r];
    hagg[(size_t)(mb + r) * COLS + n0 + 32 + nl] = c2[r];
    hagg[(size_t)(mb + r) * COLS + n0 + 48 + nl] = c3[r];
  }
}

// ---------------- output head: y = lrelu(hagg@Wl + bl) @ Wl2 + bl2 ----------------
__global__ void head_kernel(const float* __restrict__ hagg, const float* __restrict__ Wl,
                            const float* __restrict__ bl, const float* __restrict__ Wl2,
                            const float* __restrict__ bl2, float* __restrict__ y) {
  __shared__ float rowv[HH];
  __shared__ float red[HH];
  int nt = blockIdx.x, tid = threadIdx.x;           // 128 threads
  rowv[tid] = hagg[(size_t)nt * HH + tid];
  __syncthreads();
  float z = bl[tid];
  for (int i = 0; i < HH; i++) z += rowv[i] * Wl[i * HH + tid];
  red[tid] = lrelu(z) * Wl2[tid];
  __syncthreads();
  for (int s = 64; s > 0; s >>= 1) { if (tid < s) red[tid] += red[tid + s]; __syncthreads(); }
  if (tid == 0) y[nt] = red[0] + bl2[0];
}

extern "C" void kernel_launch(void* const* d_in, const int* in_sizes, int n_in,
                              void* d_out, int out_size, void* d_ws, size_t ws_size,
                              hipStream_t stream) {
  // --- inputs (setup_inputs dict order) ---
  const float* dyn0  = (const float*)d_in[0];
  const float* dw1_0 = (const float*)d_in[1];
  const float* db1_0 = (const float*)d_in[2];
  const float* dw2_0 = (const float*)d_in[3];
  const float* db2_0 = (const float*)d_in[4];
  const float* dyn1  = (const float*)d_in[5];
  const float* dw1_1 = (const float*)d_in[6];
  const float* db1_1 = (const float*)d_in[7];
  const float* dw2_1 = (const float*)d_in[8];
  const float* db2_1 = (const float*)d_in[9];
  const float* dyn2  = (const float*)d_in[10];
  const float* dw1_2 = (const float*)d_in[11];
  const float* db1_2 = (const float*)d_in[12];
  const float* dw2_2 = (const float*)d_in[13];
  const float* db2_2 = (const float*)d_in[14];
  const float* st0   = (const float*)d_in[15];
  const float* w_0   = (const float*)d_in[16];
  const float* a_0   = (const float*)d_in[17];
  const float* st1   = (const float*)d_in[18];
  const float* w_1   = (const float*)d_in[19];
  const float* a_1   = (const float*)d_in[20];
  const float* dist  = (const float*)d_in[21];
  const float* Wdis  = (const float*)d_in[22];
  const float* adis  = (const float*)d_in[23];
  const float* Wih   = (const float*)d_in[24];
  const float* Whh   = (const float*)d_in[25];
  const float* bih   = (const float*)d_in[26];
  const float* bhh   = (const float*)d_in[27];
  const float* Wl    = (const float*)d_in[28];
  const float* bl    = (const float*)d_in[29];
  const float* Wl2   = (const float*)d_in[30];
  const float* bl2   = (const float*)d_in[31];
  float* y = (float*)d_out;

  // --- workspace carve (~131 MB total) ---
  char* p = (char*)d_ws;
  auto carve = [&](size_t bytes) -> char* {
    char* q = p; p += (bytes + 255) & ~(size_t)255; return q;
  };
  _Float16* Adis    = (_Float16*)carve((size_t)MPAD * KPAD * 2);    //  8.3 MB
  _Float16* hT      = (_Float16*)carve((size_t)COLS * KPAD * 2);    // 33.0 MB
  _Float16* xh      = (_Float16*)carve((size_t)NN * TT * FF * 2);   // 16.4 MB
  _Float16* WihH    = (_Float16*)carve((size_t)G4 * FF * 2);
  _Float16* WhhH    = (_Float16*)carve((size_t)G4 * HH * 2);
  float*    bsum    = (float*)carve((size_t)G4 * 4);
  float*    c_state = (float*)carve((size_t)NN * HH * 4);
  _Float16* h16     = (_Float16*)carve((size_t)NN * HH * 2);
  float*    Gm      = (float*)carve((size_t)NN * G4 * 4);           //  4.1 MB
  float*    hagg    = (float*)carve((size_t)MPAD * COLS * 4);       // 67.1 MB
  float*    s0a     = (float*)carve((size_t)NN * 4);
  float*    s0b     = (float*)carve((size_t)NN * 4);
  float*    s1a     = (float*)carve((size_t)NN * 4);
  float*    s1b     = (float*)carve((size_t)NN * 4);
  (void)ws_size; (void)in_sizes; (void)n_in; (void)out_size;

  // --- pipeline ---
  prep_kernel<<<1000, 256, 0, stream>>>(Wih, Whh, bih, bhh, WihH, WhhH, bsum, c_state, h16, hT, Adis);
  score_kernel<<<(NN + 255) / 256, 256, 0, stream>>>(st0, w_0, a_0, st1, w_1, a_1, s0a, s0b, s1a, s1b);
  attn_kernel<<<NN, 256, 0, stream>>>(s0a, s0b, s1a, s1b, dist, Wdis, adis, Adis);
  gating_kernel<<<NN * TT, 128, 0, stream>>>(dyn0, dw1_0, db1_0, dw2_0, db2_0,
                                             dyn1, dw1_1, db1_1, dw2_1, db2_1,
                                             dyn2, dw1_2, db1_2, dw2_2, db2_2, xh);
  for (int t = 0; t < TT; t++) {
    lstm_gemm_kernel<<<(125 * 8) / 4, 128, 0, stream>>>(xh, t, h16, WihH, WhhH, bsum, Gm);
    lstm_gate_kernel<<<(NN * HH + 255) / 256, 256, 0, stream>>>(Gm, c_state, h16, hT, t);
  }
  agg_gemm_kernel<<<(MPAD / 64) * (COLS / 64), 128, 0, stream>>>(Adis, hT, hagg);
  head_kernel<<<NN * TT, 128, 0, stream>>>(hagg, Wl, bl, Wl2, bl2, y);
}
